// CausalMultiHeadAttention_43731357008712
// MI455X (gfx1250) — matrix operand order repaired and checked by reading, not compiled, not run on hardware
//
#include <hip/hip_runtime.h>
#include <hip/hip_bf16.h>

#define BB 2
#define SS 2048
#define DM 1024
#define HH 16
#define HD 64
#define MTOT (BB * SS) /* 4096 */

typedef __bf16 bf16_t;
typedef __attribute__((ext_vector_type(16))) __bf16 v16bf;
typedef __attribute__((ext_vector_type(8)))  __bf16 v8bf;
typedef __attribute__((ext_vector_type(8)))  float  v8f;

typedef unsigned int u32;
typedef __attribute__((ext_vector_type(4))) unsigned int u32x4;
typedef __attribute__((ext_vector_type(4))) int          i32x4;
typedef __attribute__((ext_vector_type(8))) int          i32x8;

#if defined(__HIP_DEVICE_COMPILE__) && __has_builtin(__builtin_amdgcn_tensor_load_to_lds) && __has_builtin(__builtin_amdgcn_s_wait_tensorcnt)
#define HAVE_TDM 1
#else
#define HAVE_TDM 0
#endif

static __device__ __forceinline__ v16bf join8(v8bf lo, v8bf hi) {
    union { v16bf v; v8bf h[2]; } u;
    u.h[0] = lo; u.h[1] = hi;
    return u.v;
}

// ---------------------------------------------------------------------------
// fp32 -> bf16 conversion
// ---------------------------------------------------------------------------
__global__ void cvt_f32_bf16(const float* __restrict__ src,
                             bf16_t* __restrict__ dst, int n) {
    int i = blockIdx.x * blockDim.x + threadIdx.x;
    if (i < n) dst[i] = (bf16_t)src[i];
}

// ---------------------------------------------------------------------------
// C[m,n] = sum_k A[m,k] * W[n,k]   (A: M x 1024, W: 1024 x 1024, row-major)
// Wave computes a 16x64 strip; 8 waves (4 M x 2 N) per 256-thread block.
// Optional fused RoPE epilogue (Q/K projections); optional fp32 output.
// ---------------------------------------------------------------------------
__global__ __launch_bounds__(256)
void gemm_nt_bf16(const bf16_t* __restrict__ A,
                  const bf16_t* __restrict__ W,
                  bf16_t* __restrict__ Cbf,
                  float* __restrict__ Cf,
                  int rope_mode,
                  const int* __restrict__ use_rope,
                  const int* __restrict__ pos,
                  const float* __restrict__ cosp,
                  const float* __restrict__ sinp) {
    const int lane = threadIdx.x & 31;
    const int wid  = threadIdx.x >> 5;
    const int g    = lane >> 4;        // half-wave group (K-chunk selector)
    const int nl   = lane & 15;
    const int m0   = blockIdx.y * 64 + (wid & 3) * 16;
    const int n0   = blockIdx.x * 128 + (wid >> 2) * 64;

    v8f acc[4] = {};

    const bf16_t* arow = A + (size_t)(m0 + nl) * DM;
    for (int k0 = 0; k0 < DM; k0 += 32) {
        // A-frag (16x32 bf16): lanes 0-15 hold K [0..15]; 16-31 hold [16..31] (contiguous)
        v16bf a = *(const v16bf*)(arow + k0 + g * 16);
#pragma unroll
        for (int t = 0; t < 4; ++t) {
            // B-frag (32x16): lane n holds W[n, k0 + g*16 .. +16] (contiguous)
            const bf16_t* wrow = W + (size_t)(n0 + t * 16 + nl) * DM + k0 + g * 16;
            v16bf b = *(const v16bf*)wrow;
            acc[t] = __builtin_amdgcn_wmma_f32_16x16x32_bf16(
                false, a, false, b, (short)0, acc[t], false, false);
        }
    }

    const bool do_rope = (rope_mode != 0) && (*use_rope != 0);
#pragma unroll
    for (int t = 0; t < 4; ++t) {
#pragma unroll
        for (int v = 0; v < 8; ++v) {
            float val = acc[t][v];
            int m = m0 + v + 8 * g;            // C layout: lanes16-31 hold M+8
            int n = n0 + t * 16 + nl;
            if (do_rope) {
                int sidx = m & (SS - 1);
                int p    = pos[sidx];
                int fi   = (n & (HD - 1)) >> 1;
                float c  = cosp[p * (HD / 2) + fi];
                float sn = sinp[p * (HD / 2) + fi];
                float partner = __shfl_xor(val, 1, 32); // even<->odd feature pair
                val = ((lane & 1) == 0) ? (val * c - partner * sn)
                                        : (val * c + partner * sn);
            }
            if (Cf) Cf[(size_t)m * DM + n] = val;
            else    Cbf[(size_t)m * DM + n] = (bf16_t)val;
        }
    }
}

// ---------------------------------------------------------------------------
// Causal flash attention over bf16 Q/K/V (RoPE already applied), fp32 softmax.
// Block = 4 waves = 64 q rows of one (b,h). The 32x64 V tile for each kv chunk
// is staged into LDS once per block via the Tensor Data Mover and shared by
// all 4 waves; waves beyond their causal end skip compute but keep barriers.
// ---------------------------------------------------------------------------
__global__ __launch_bounds__(128)
void flash_attn(const bf16_t* __restrict__ Q,
                const bf16_t* __restrict__ K,
                const bf16_t* __restrict__ V,
                bf16_t* __restrict__ Y) {
    __shared__ __align__(16) bf16_t vtile[32][64];   // 4KB V tile (TDM target)
    __shared__ __align__(16) bf16_t plds[4][16][40]; // P staging, padded stride

    const int lane = threadIdx.x & 31;
    const int wid  = threadIdx.x >> 5;
    const int g    = lane >> 4;
    const int nl   = lane & 15;
    const int bh   = blockIdx.y;
    const int b    = bh / HH;
    const int h    = bh % HH;
    const int q0r  = blockIdx.x * 64 + wid * 16;
    const size_t base = (size_t)b * SS * DM + (size_t)h * HD;

    // Q A-frags for d = 0..31 and 32..63, kept in registers for the whole loop
    const bf16_t* qrow = Q + base + (size_t)(q0r + nl) * DM;
    v16bf qa0 = *(const v16bf*)(qrow + g * 16);
    v16bf qa1 = *(const v16bf*)(qrow + 32 + g * 16);

    v8f o[4] = {};
    float mrun[8], lrun[8];
#pragma unroll
    for (int v = 0; v < 8; ++v) { mrun[v] = -1e30f; lrun[v] = 0.0f; }

    const int qend      = q0r + 15;
    const int q_blk_end = blockIdx.x * 64 + 63;

#if HAVE_TDM
    const u32 lds_off = (u32)(uintptr_t)&vtile[0][0]; // LDS aperture: addr[31:0]
#endif

    for (int kb = 0; kb <= q_blk_end; kb += 32) {
        __syncthreads(); // previous tile fully consumed before overwrite

#if HAVE_TDM
        if (wid == 0) {
            // TDM 2D descriptor: 64x32 bf16 tile, row stride 1024 elements
            unsigned long long gaddr =
                (unsigned long long)(uintptr_t)(V + base + (size_t)kb * DM);
            u32x4 g0 = { 1u,                                    // count=1
                         lds_off,                               // lds_addr
                         (u32)gaddr,                            // global_addr[31:0]
                         (u32)((gaddr >> 32) & 0x01FFFFFFu) | 0x80000000u }; // addr[56:32] | type=2
            i32x8 g1 = { (int)(1u << 16),   // data_size = 1 (2 bytes)
                         (int)(64u << 16),  // tensor_dim0 = 64 (bits 79:48)
                         (int)(32u << 16),  // tensor_dim1 = 32 (bits 111:80)
                         (int)(64u << 16),  // tile_dim0 = 64 (bits 127:112)
                         32,                // tile_dim1 = 32 (bits 143:128)
                         1024,              // tensor_dim0_stride (bits 207:160)
                         0, 0 };
            i32x4 gz = { 0, 0, 0, 0 };
#if __clang_major__ >= 23
            i32x8 gz8 = { 0, 0, 0, 0, 0, 0, 0, 0 };
            __builtin_amdgcn_tensor_load_to_lds(g0, g1, gz, gz, gz8, 0);
#else
            __builtin_amdgcn_tensor_load_to_lds(g0, g1, gz, gz, 0);
#endif
            __builtin_amdgcn_s_wait_tensorcnt((short)0);
        }
#else
        // fallback: cooperative global -> LDS staging of the V tile
        for (int c = threadIdx.x; c < 256; c += 128) {
            int r = c >> 3, c8 = (c & 7) * 8;
            *(v8bf*)&vtile[r][c8] =
                *(const v8bf*)(V + base + (size_t)(kb + r) * DM + c8);
        }
#endif
        __syncthreads(); // V tile visible to all waves

        if (kb > qend) continue; // fully-masked chunk for this wave: barriers only

        // prefetch next K tile while this chunk computes
        if (kb + 32 <= qend)
            __builtin_prefetch(K + base + (size_t)(kb + 32 + nl) * DM, 0, 3);

        // S tiles: S[m, j0+n] = Q . K^T, two 16-col tiles per 32-key chunk
        v8f s[2];
#pragma unroll
        for (int t = 0; t < 2; ++t) {
            const bf16_t* krow = K + base + (size_t)(kb + t * 16 + nl) * DM;
            v16bf b0 = *(const v16bf*)(krow + g * 16);
            v16bf b1 = *(const v16bf*)(krow + 32 + g * 16);
            v8f a = {};
            a = __builtin_amdgcn_wmma_f32_16x16x32_bf16(false, qa0, false, b0,
                                                        (short)0, a, false, false);
            a = __builtin_amdgcn_wmma_f32_16x16x32_bf16(false, qa1, false, b1,
                                                        (short)0, a, false, false);
            s[t] = a;
        }

        // scale + causal mask + online softmax (rows split across 16-lane halves)
        float p0[8], p1[8];
#pragma unroll
        for (int v = 0; v < 8; ++v) {
            int m  = q0r + v + 8 * g;
            float x0 = s[0][v] * 0.125f;               // 1/sqrt(64)
            float x1 = s[1][v] * 0.125f;
            if (kb + nl > m)      x0 = -1e30f;
            if (kb + 16 + nl > m) x1 = -1e30f;
            float mx = fmaxf(x0, x1);
#pragma unroll
            for (int d = 1; d < 16; d <<= 1) mx = fmaxf(mx, __shfl_xor(mx, d, 32));
            float mnew  = fmaxf(mrun[v], mx);
            float alpha = __expf(mrun[v] - mnew);
            float e0 = __expf(x0 - mnew);
            float e1 = __expf(x1 - mnew);
            float rs = e0 + e1;
#pragma unroll
            for (int d = 1; d < 16; d <<= 1) rs += __shfl_xor(rs, d, 32);
            lrun[v] = lrun[v] * alpha + rs;
            mrun[v] = mnew;
#pragma unroll
            for (int t = 0; t < 4; ++t) o[t][v] *= alpha;
            p0[v] = e0; p1[v] = e1;
        }

        // C-layout -> A-layout relayout of P through LDS (same wave: DS in-order)
#pragma unroll
        for (int v = 0; v < 8; ++v) {
            int m = v + 8 * g;
            plds[wid][m][nl]      = (bf16_t)p0[v];
            plds[wid][m][16 + nl] = (bf16_t)p1[v];
        }
        v16bf pa = join8(*(const v8bf*)&plds[wid][nl][g * 16],
                         *(const v8bf*)&plds[wid][nl][g * 16 + 8]);

        // O += P(16x32) x V(32x64): B-frag lane n reads a column of the LDS tile
#pragma unroll
        for (int t = 0; t < 4; ++t) {
            const bf16_t* vp = &vtile[g * 16][t * 16 + nl];
            v16bf vb;
#pragma unroll
            for (int i = 0; i < 16; ++i) vb[i] = vp[i * 64];
            o[t] = __builtin_amdgcn_wmma_f32_16x16x32_bf16(
                false, pa, false, vb, (short)0, o[t], false, false);
        }
    }

    // normalize and write Y[b, s, h*64 + n] as bf16
#pragma unroll
    for (int t = 0; t < 4; ++t) {
#pragma unroll
        for (int v = 0; v < 8; ++v) {
            int m = q0r + v + 8 * g;
            float val = o[t][v] / lrun[v];
            Y[base + (size_t)m * DM + t * 16 + nl] = (bf16_t)val;
        }
    }
}

// ---------------------------------------------------------------------------
extern "C" void kernel_launch(void* const* d_in, const int* in_sizes, int n_in,
                              void* d_out, int out_size, void* d_ws, size_t ws_size,
                              hipStream_t stream) {
    (void)in_sizes; (void)n_in; (void)out_size; (void)ws_size;
    const float* x      = (const float*)d_in[0];
    const int*   pos    = (const int*)d_in[1];
    const int*   urope  = (const int*)d_in[2];
    const float* Wq     = (const float*)d_in[3];
    const float* Wk     = (const float*)d_in[4];
    const float* Wv     = (const float*)d_in[5];
    const float* Wo     = (const float*)d_in[6];
    const float* cosp   = (const float*)d_in[7];
    const float* sinp   = (const float*)d_in[8];
    float*       out    = (float*)d_out;

    char* ws = (char*)d_ws;
    size_t off = 0;
    auto alloc = [&](size_t bytes) {
        char* p = ws + off;
        off += (bytes + 255) & ~(size_t)255;
        return p;
    };
    bf16_t* xb  = (bf16_t*)alloc((size_t)MTOT * DM * 2);
    bf16_t* wqb = (bf16_t*)alloc((size_t)DM * DM * 2);
    bf16_t* wkb = (bf16_t*)alloc((size_t)DM * DM * 2);
    bf16_t* wvb = (bf16_t*)alloc((size_t)DM * DM * 2);
    bf16_t* wob = (bf16_t*)alloc((size_t)DM * DM * 2);
    bf16_t* Qb  = (bf16_t*)alloc((size_t)MTOT * DM * 2);
    bf16_t* Kb  = (bf16_t*)alloc((size_t)MTOT * DM * 2);
    bf16_t* Vb  = (bf16_t*)alloc((size_t)MTOT * DM * 2);
    bf16_t* Yb  = (bf16_t*)alloc((size_t)MTOT * DM * 2);

    const int nx = MTOT * DM;
    const int nw = DM * DM;
    cvt_f32_bf16<<<(nx + 255) / 256, 256, 0, stream>>>(x,  xb,  nx);
    cvt_f32_bf16<<<(nw + 255) / 256, 256, 0, stream>>>(Wq, wqb, nw);
    cvt_f32_bf16<<<(nw + 255) / 256, 256, 0, stream>>>(Wk, wkb, nw);
    cvt_f32_bf16<<<(nw + 255) / 256, 256, 0, stream>>>(Wv, wvb, nw);
    cvt_f32_bf16<<<(nw + 255) / 256, 256, 0, stream>>>(Wo, wob, nw);

    dim3 gg(DM / 128, MTOT / 64);
    gemm_nt_bf16<<<gg, 256, 0, stream>>>(xb, wqb, Qb, nullptr, 1, urope, pos, cosp, sinp);
    gemm_nt_bf16<<<gg, 256, 0, stream>>>(xb, wkb, Kb, nullptr, 1, urope, pos, cosp, sinp);
    gemm_nt_bf16<<<gg, 256, 0, stream>>>(xb, wvb, Vb, nullptr, 0, urope, pos, cosp, sinp);

    dim3 ga(SS / 64, BB * HH);
    flash_attn<<<ga, 128, 0, stream>>>(Qb, Kb, Vb, Yb);

    gemm_nt_bf16<<<gg, 256, 0, stream>>>(Yb, wob, nullptr, out, 0, urope, pos, cosp, sinp);
}
